// DielectricGAT_23038204576148
// MI455X (gfx1250) — matrix operand order repaired and checked by reading, not compiled, not run on hardware
//
#include <hip/hip_runtime.h>
#include <hip/hip_bf16.h>
#include <stdint.h>

typedef __attribute__((ext_vector_type(16))) _Float16 v16h;
typedef __attribute__((ext_vector_type(8)))  _Float16 v8h;
typedef __attribute__((ext_vector_type(8)))  float    v8f;

union HV16 { v16h v; v8h h[2]; };

#define TPB 256

// -------------------- utility kernels --------------------

__global__ void k_fill_f32(float* __restrict__ p, long long n, float v) {
    long long i = (long long)blockIdx.x * blockDim.x + threadIdx.x;
    if (i < n) p[i] = v;
}

__global__ void k_f32_to_f16(const float* __restrict__ in, _Float16* __restrict__ out,
                             long long n) {
    long long i = (long long)blockIdx.x * blockDim.x + threadIdx.x;
    if (i < n) out[i] = (_Float16)in[i];
}

// out[col*rows + row] = in[row*cols + col]   (rows = K, cols = Ncols)
__global__ void k_transpose_f16(const float* __restrict__ in, _Float16* __restrict__ out,
                                int rows, int cols) {
    long long t = (long long)blockIdx.x * blockDim.x + threadIdx.x;
    if (t >= (long long)rows * cols) return;
    int col = (int)(t / rows);
    int row = (int)(t % rows);
    out[t] = (_Float16)in[(long long)row * cols + col];
}

// -------------------- WMMA GEMM: D[M][Ncols] = A[M][K] * B[K][Ncols] --------------------
// A is f16 row-major; BT is f16 [Ncols][K] (i.e. B transposed); fp32 accumulate.
// One wave computes one 16x16 output tile; K stepped by 32 via v_wmma_f32_16x16x32_f16.
__global__ __launch_bounds__(TPB) void k_wmma_gemm(
    const _Float16* __restrict__ A, const _Float16* __restrict__ BT,
    float* __restrict__ D, int M, int K, int Ncols)
{
    int gwave  = (int)(((long long)blockIdx.x * blockDim.x + threadIdx.x) >> 5);
    int lane   = threadIdx.x & 31;
    int tilesN = Ncols >> 4;
    int tilesM = M >> 4;
    if (gwave >= tilesM * tilesN) return;          // wave-uniform: EXEC stays all-1s
    int tileM  = gwave / tilesN;
    int tileN  = gwave - tileM * tilesN;
    int lane16 = lane & 15;
    int khi    = lane >> 4;                        // 0: K-low half, 1: K-high half

    const _Float16* arow = A  + (long long)(tileM * 16 + lane16) * K;
    const _Float16* bcol = BT + (long long)(tileN * 16 + lane16) * K;

    v8f acc = {};
    for (int k0 = 0; k0 < K; k0 += 32) {
        HV16 a, b;
        // A 16x32 f16 layout: halves 0..7 -> K = k0 + khi*8 + [0..7]
        //                     halves 8..15 -> K = k0 + khi*8 + 16 + [0..7]
        a.h[0] = *(const v8h*)(arow + k0 + khi * 8);
        a.h[1] = *(const v8h*)(arow + k0 + khi * 8 + 16);
        // B 32x16 f16 layout: same K mapping as A (khi*8 + [0..7], khi*8 + 16 + [0..7])
        b.h[0] = *(const v8h*)(bcol + k0 + khi * 8);
        b.h[1] = *(const v8h*)(bcol + k0 + khi * 8 + 16);
        acc = __builtin_amdgcn_wmma_f32_16x16x32_f16(
                  false, a.v, false, b.v, (short)0, acc, false, false);
    }

    // D 16x16 f32: lane<16 -> rows 0..7 col=lane ; lane>=16 -> rows 8..15 col=lane-16
    int orow0 = tileM * 16 + khi * 8;
    int ocol  = tileN * 16 + lane16;
    #pragma unroll
    for (int r = 0; r < 8; ++r)
        D[(long long)(orow0 + r) * Ncols + ocol] = acc[r];
}

// -------------------- attention --------------------

__device__ inline void atomicMaxF(float* addr, float v) {
    // valid for mixed signs with init = -1e30f
    if (v >= 0.0f) atomicMax((int*)addr, __float_as_int(v));
    else           atomicMin((unsigned int*)addr, __float_as_uint(v));
}

// per (node, head): a_s = <h, att_src>, a_d = <h, att_dst>  (C = 64)
__global__ void k_node_attn(const float* __restrict__ h,
                            const float* __restrict__ att_s,
                            const float* __restrict__ att_d,
                            float* __restrict__ as_, float* __restrict__ ad_,
                            int n, int Hh) {
    int t = blockIdx.x * blockDim.x + threadIdx.x;
    if (t >= n * Hh) return;
    int hd = t % Hh;
    const float* hp = h + (long long)t * 64;
    const float* sp = att_s + hd * 64;
    const float* dp = att_d + hd * 64;
    float s = 0.f, d = 0.f;
    #pragma unroll 8
    for (int c = 0; c < 64; ++c) { float v = hp[c]; s += v * sp[c]; d += v * dp[c]; }
    as_[t] = s; ad_[t] = d;
}

__device__ inline void edge_nodes(const int* __restrict__ ei, int E, int e, int& s, int& d) {
    if (e < E) { s = ei[e]; d = ei[E + e]; } else { s = d = e - E; }  // self-loops appended
}

// alpha = leaky_relu(a_s[src] + a_d[dst]); segment max into amax[dst]
__global__ void k_edge_alpha(const int* __restrict__ ei, int E, int EE,
                             const float* __restrict__ as_, const float* __restrict__ ad_,
                             float* __restrict__ alpha, float* __restrict__ amax, int Hh) {
    int e = blockIdx.x * blockDim.x + threadIdx.x;
    if (e >= EE) return;
    int s, d; edge_nodes(ei, E, e, s, d);
    for (int hh = 0; hh < Hh; ++hh) {
        float a = as_[s * Hh + hh] + ad_[d * Hh + hh];
        a = (a > 0.f) ? a : 0.2f * a;
        alpha[(long long)e * Hh + hh] = a;
        atomicMaxF(&amax[d * Hh + hh], a);
    }
}

// ex = exp(alpha - amax[dst]) (stored over alpha); segment sum into den[dst]
__global__ void k_edge_ex(const int* __restrict__ ei, int E, int EE,
                          float* __restrict__ alpha, const float* __restrict__ amax,
                          float* __restrict__ den, int Hh) {
    int e = blockIdx.x * blockDim.x + threadIdx.x;
    if (e >= EE) return;
    int s, d; edge_nodes(ei, E, e, s, d);
    for (int hh = 0; hh < Hh; ++hh) {
        long long idx = (long long)e * Hh + hh;
        float ex = __expf(alpha[idx] - amax[d * Hh + hh]);
        alpha[idx] = ex;
        atomicAdd(&den[d * Hh + hh], ex);
    }
}

// out[dst,h,:] += h[src,h,:] * ex/(den[dst,h]+1e-16)
__global__ void k_edge_scatter(const int* __restrict__ ei, int E, int EE,
                               const float* __restrict__ hsrc,
                               const float* __restrict__ ex, const float* __restrict__ den,
                               float* __restrict__ out, int Hh) {
    long long t = (long long)blockIdx.x * blockDim.x + threadIdx.x;
    if (t >= (long long)EE * Hh) return;
    int e  = (int)(t / Hh);
    int hh = (int)(t % Hh);
    int s, d; edge_nodes(ei, E, e, s, d);
    float w = ex[t] / (den[d * Hh + hh] + 1e-16f);
    const float4* hp = (const float4*)(hsrc + ((long long)s * Hh + hh) * 64);
    float* op = out + ((long long)d * Hh + hh) * 64;
    #pragma unroll
    for (int c4 = 0; c4 < 16; ++c4) {
        float4 v = hp[c4];
        atomicAdd(op + c4 * 4 + 0, v.x * w);
        atomicAdd(op + c4 * 4 + 1, v.y * w);
        atomicAdd(op + c4 * 4 + 2, v.z * w);
        atomicAdd(op + c4 * 4 + 3, v.w * w);
    }
}

// y = relu(out + bias[col]); optionally write f16 copy for next WMMA GEMM
__global__ void k_bias_relu_f16(const float* __restrict__ in, const float* __restrict__ bias,
                                _Float16* __restrict__ outh, long long n, int ncols) {
    long long t = (long long)blockIdx.x * blockDim.x + threadIdx.x;
    if (t >= n) return;
    float v = in[t] + bias[(int)(t % ncols)];
    outh[t] = (_Float16)fmaxf(v, 0.f);
}

__global__ void k_bias_relu_inplace(float* __restrict__ io, const float* __restrict__ bias,
                                    long long n, int ncols) {
    long long t = (long long)blockIdx.x * blockDim.x + threadIdx.x;
    if (t >= n) return;
    io[t] = fmaxf(io[t] + bias[(int)(t % ncols)], 0.f);
}

// -------------------- pooling + FC --------------------

__global__ void k_pool(const float* __restrict__ y, const int* __restrict__ batch,
                       float* __restrict__ pool, float* __restrict__ cnt, int n) {
    long long t = (long long)blockIdx.x * blockDim.x + threadIdx.x;
    if (t >= (long long)n * 64) return;
    int node = (int)(t >> 6), c = (int)(t & 63);
    int b = batch[node];
    atomicAdd(&pool[(long long)b * 64 + c], y[t]);
    if (c == 0) atomicAdd(&cnt[b], 1.0f);
}

__global__ void k_fc(const float* __restrict__ pool, const float* __restrict__ cnt,
                     const float* __restrict__ fcw, const float* __restrict__ fcb,
                     float* __restrict__ out, int G) {
    int g = blockIdx.x * blockDim.x + threadIdx.x;
    if (g >= G) return;
    float inv = 1.0f / fmaxf(cnt[g], 1.0f);
    float acc = 0.f;
    #pragma unroll 8
    for (int c = 0; c < 64; ++c) acc += pool[(long long)g * 64 + c] * inv * fcw[c];
    out[g] = acc + fcb[0];
}

// -------------------- host launcher --------------------

static inline dim3 grid1(long long n) { return dim3((unsigned)((n + TPB - 1) / TPB)); }

extern "C" void kernel_launch(void* const* d_in, const int* in_sizes, int n_in,
                              void* d_out, int out_size, void* d_ws, size_t ws_size,
                              hipStream_t stream) {
    const float* x        = (const float*)d_in[0];
    const float* W1       = (const float*)d_in[1];
    const float* att_src1 = (const float*)d_in[2];
    const float* att_dst1 = (const float*)d_in[3];
    const float* b1       = (const float*)d_in[4];
    const float* W2       = (const float*)d_in[5];
    const float* att_src2 = (const float*)d_in[6];
    const float* att_dst2 = (const float*)d_in[7];
    const float* b2       = (const float*)d_in[8];
    const float* fc_w     = (const float*)d_in[9];
    const float* fc_b     = (const float*)d_in[10];
    const int*   ei       = (const int*)d_in[11];
    const int*   batch    = (const int*)d_in[12];

    const int FIN = 128, H = 4, C = 64, HC = H * C;   // 256
    const int N  = in_sizes[0] / FIN;
    const int E  = in_sizes[11] / 2;
    const int EE = E + N;                              // + self loops
    const int G  = out_size;
    float* out = (float*)d_out;

    // ---- carve workspace ----
    uintptr_t base = (uintptr_t)d_ws;
    size_t off = 0;
    auto alloc = [&](size_t bytes) -> void* {
        off = (off + 255) & ~(size_t)255;
        void* p = (void*)(base + off);
        off += bytes;
        return p;
    };
    _Float16* xh    = (_Float16*)alloc((size_t)N * FIN * 2);
    _Float16* w1t   = (_Float16*)alloc((size_t)HC * FIN * 2);
    float*    h1    = (float*)   alloc((size_t)N * HC * 4);
    float*    as1   = (float*)   alloc((size_t)N * H * 4);
    float*    ad1   = (float*)   alloc((size_t)N * H * 4);
    float*    amax1 = (float*)   alloc((size_t)N * H * 4);
    float*    den1  = (float*)   alloc((size_t)N * H * 4);
    float*    alp1  = (float*)   alloc((size_t)EE * H * 4);
    float*    out1  = (float*)   alloc((size_t)N * HC * 4);
    _Float16* y1h   = (_Float16*)alloc((size_t)N * HC * 2);
    _Float16* w2t   = (_Float16*)alloc((size_t)C * HC * 2);
    float*    h2    = (float*)   alloc((size_t)N * C * 4);
    float*    as2   = (float*)   alloc((size_t)N * 4);
    float*    ad2   = (float*)   alloc((size_t)N * 4);
    float*    amax2 = (float*)   alloc((size_t)N * 4);
    float*    den2  = (float*)   alloc((size_t)N * 4);
    float*    alp2  = (float*)   alloc((size_t)EE * 4);
    float*    out2  = (float*)   alloc((size_t)N * C * 4);
    float*    pool  = (float*)   alloc((size_t)G * C * 4);
    float*    cnt   = (float*)   alloc((size_t)G * 4);

    // ---- init accumulators (ws is poisoned; re-init every call) ----
    k_fill_f32<<<grid1((long long)N * H), TPB, 0, stream>>>(amax1, (long long)N * H, -1e30f);
    k_fill_f32<<<grid1((long long)N * H), TPB, 0, stream>>>(den1,  (long long)N * H, 0.f);
    k_fill_f32<<<grid1((long long)N * HC), TPB, 0, stream>>>(out1, (long long)N * HC, 0.f);
    k_fill_f32<<<grid1(N), TPB, 0, stream>>>(amax2, N, -1e30f);
    k_fill_f32<<<grid1(N), TPB, 0, stream>>>(den2,  N, 0.f);
    k_fill_f32<<<grid1((long long)N * C), TPB, 0, stream>>>(out2, (long long)N * C, 0.f);
    k_fill_f32<<<grid1((long long)G * C), TPB, 0, stream>>>(pool, (long long)G * C, 0.f);
    k_fill_f32<<<grid1(G), TPB, 0, stream>>>(cnt, G, 0.f);

    // ---- f16 operands for WMMA ----
    k_f32_to_f16<<<grid1((long long)N * FIN), TPB, 0, stream>>>(x, xh, (long long)N * FIN);
    k_transpose_f16<<<grid1((long long)FIN * HC), TPB, 0, stream>>>(W1, w1t, FIN, HC);
    k_transpose_f16<<<grid1((long long)HC * C), TPB, 0, stream>>>(W2, w2t, HC, C);

    // ---- layer 1: GEMM (WMMA), attention, aggregation ----
    long long tiles1 = (long long)(N / 16) * (HC / 16);
    k_wmma_gemm<<<grid1(tiles1 * 32), TPB, 0, stream>>>(xh, w1t, h1, N, FIN, HC);
    k_node_attn<<<grid1((long long)N * H), TPB, 0, stream>>>(h1, att_src1, att_dst1, as1, ad1, N, H);
    k_edge_alpha<<<grid1(EE), TPB, 0, stream>>>(ei, E, EE, as1, ad1, alp1, amax1, H);
    k_edge_ex<<<grid1(EE), TPB, 0, stream>>>(ei, E, EE, alp1, amax1, den1, H);
    k_edge_scatter<<<grid1((long long)EE * H), TPB, 0, stream>>>(ei, E, EE, h1, alp1, den1, out1, H);
    k_bias_relu_f16<<<grid1((long long)N * HC), TPB, 0, stream>>>(out1, b1, y1h, (long long)N * HC, HC);

    // ---- layer 2: GEMM (WMMA), attention (H=1), aggregation ----
    long long tiles2 = (long long)(N / 16) * (C / 16);
    k_wmma_gemm<<<grid1(tiles2 * 32), TPB, 0, stream>>>(y1h, w2t, h2, N, HC, C);
    k_node_attn<<<grid1(N), TPB, 0, stream>>>(h2, att_src2, att_dst2, as2, ad2, N, 1);
    k_edge_alpha<<<grid1(EE), TPB, 0, stream>>>(ei, E, EE, as2, ad2, alp2, amax2, 1);
    k_edge_ex<<<grid1(EE), TPB, 0, stream>>>(ei, E, EE, alp2, amax2, den2, 1);
    k_edge_scatter<<<grid1(EE), TPB, 0, stream>>>(ei, E, EE, h2, alp2, den2, out2, 1);
    k_bias_relu_inplace<<<grid1((long long)N * C), TPB, 0, stream>>>(out2, b2, (long long)N * C, C);

    // ---- mean pool + FC ----
    k_pool<<<grid1((long long)N * C), TPB, 0, stream>>>(out2, batch, pool, cnt, N);
    k_fc<<<grid1(G), TPB, 0, stream>>>(pool, cnt, fc_w, fc_b, out, G);
}
